// CombinedModel_72919954751845
// MI455X (gfx1250) — compile-verified, matrix operand order checked by reading, not run on hardware
//
#include <hip/hip_runtime.h>
#include <hip/hip_bf16.h>
#include <math.h>

typedef __attribute__((ext_vector_type(16))) _Float16 v16h;
typedef __attribute__((ext_vector_type(8)))  _Float16 v8h;
typedef __attribute__((ext_vector_type(8)))  float    v8f;

#define NNODES 32768

// CDNA5 async global->LDS path (guarded: falls back to VGPR staging if absent)
#if __has_builtin(__builtin_amdgcn_global_load_async_to_lds_b128) && \
    __has_builtin(__builtin_amdgcn_s_wait_asynccnt)
#define HAVE_ASYNC_LDS 1
#define ASG __attribute__((address_space(1)))
#define ASL __attribute__((address_space(3)))
typedef __attribute__((__vector_size__(4 * sizeof(int)))) int i32x4;
#endif

// ---------------------------------------------------------------------------
// Weight packing: pack fp32 W[K,Fo] into f16 B-fragment order for
// v_wmma_f32_16x16x32_f16.  Fragment layout (ISA 7.12.2, 16-bit operands):
//   half h -> k = (h&7) + 8*(lane>=16) + 16*(h>>3),  n = lane&15
// Packed index = ((kk*ntn + nt)*32 + lane)*16 + h   (contiguous v16h per lane)
// ---------------------------------------------------------------------------
__global__ void pack_dense_kernel(const float* __restrict__ W, int K, int Fo,
                                  _Float16* __restrict__ Wp, int ntn, int ksteps)
{
    int tid = blockIdx.x * blockDim.x + threadIdx.x;
    int total = ksteps * ntn * 512;
    if (tid >= total) return;
    int h    = tid & 15;
    int lane = (tid >> 4) & 31;
    int tile = tid >> 9;
    int nt = tile % ntn;
    int kk = tile / ntn;
    int n = nt * 16 + (lane & 15);
    int k = kk * 32 + (h & 7) + ((lane >> 4) << 3) + ((h >> 3) << 4);
    float v = (k < K && n < Fo) ? W[(size_t)k * Fo + n] : 0.f;
    Wp[tid] = (_Float16)v;
}

// Pack LSTM weight w[4H, F] (torch gate order i,f,g,o) transposed into
// [F_pad=32, 128] with GATE-INTERLEAVED columns:
//   packed col n: gate g = (n>>4)&3, hidden j = ((n>>6)<<4) + (n&15)
// i.e. column tiles = [i0 f0 g0 o0 | i1 f1 g1 o1]  (16-wide hidden groups)
// so the LSTM kernel only needs 4 live accumulators at a time.
__global__ void pack_lstm_kernel(const float* __restrict__ W, int H, int F,
                                 _Float16* __restrict__ Wp)
{
    int tid = blockIdx.x * blockDim.x + threadIdx.x;
    if (tid >= 4096) return;                    // ksteps=1, ntn=8
    int h    = tid & 15;
    int lane = (tid >> 4) & 31;
    int tile = tid >> 9;                        // nt (kk==0)
    int n = tile * 16 + (lane & 15);
    int g = (n >> 4) & 3;
    int j = ((n >> 6) << 4) + (n & 15);
    int k = (h & 7) + ((lane >> 4) << 3) + ((h >> 3) << 4);
    float v = (j < H && k < F) ? W[(size_t)(g * H + j) * F + k] : 0.f;
    Wp[tid] = (_Float16)v;
}

__global__ void pack_lstm_bias_kernel(const float* __restrict__ b, int H,
                                      float* __restrict__ bo)
{
    int n = threadIdx.x;                        // 128 threads
    int g = (n >> 4) & 3;
    int j = ((n >> 6) << 4) + (n & 15);
    bo[n] = (j < H) ? b[g * H + j] : 0.f;
}

// fp32 [N,K] -> f16 [N,Kpad] zero padded
__global__ void cvt_pad_kernel(const float* __restrict__ X, int K, int Kpad,
                               _Float16* __restrict__ Y, int total)
{
    int tid = blockIdx.x * blockDim.x + threadIdx.x;
    if (tid >= total) return;
    int n = tid / Kpad, c = tid % Kpad;
    Y[tid] = (c < K) ? (_Float16)X[(size_t)n * K + c] : (_Float16)0.f;
}

// ---------------------------------------------------------------------------
// Row-strip WMMA GEMM:  Y = act(X @ W + bias + Z)
// One wave owns a 16 x (16*NTN) output strip: the A fragment is loaded ONCE
// per K-step and reused across NTN back-to-back WMMAs (NTN accumulators live
// in VGPRs).  act: 0=none, 1=relu, 2=elu
// ---------------------------------------------------------------------------
template <int NTN>
__global__ void __launch_bounds__(32, 1)
wmma_gemm_strip_kernel(const _Float16* __restrict__ X, int ldx,
                       const _Float16* __restrict__ Wp,
                       const float* __restrict__ bias,
                       const float* __restrict__ Z, int ldz,
                       float* __restrict__ Yf32, int ldy32,
                       _Float16* __restrict__ Yf16, int ldy16,
                       int ksteps, int act)
{
    int lane = threadIdx.x;
    int mt   = blockIdx.x;
    int hi = lane >> 4;
    int lm = lane & 15;

    const _Float16* xrow = X + (size_t)(mt * 16 + lm) * ldx + hi * 8;
    const _Float16* wp   = Wp + (size_t)lane * 16;

    v8f acc[NTN];
#pragma unroll
    for (int nt = 0; nt < NTN; ++nt) acc[nt] = (v8f){};

    for (int kk = 0; kk < ksteps; ++kk) {
        if (kk + 1 < ksteps)
            __builtin_prefetch(xrow + (kk + 1) * 32, 0, 3);   // global_prefetch_b8
        v8h c0 = *(const v8h*)(xrow + kk * 32);        // halves 0..7  : k = hi*8+0..7
        v8h c1 = *(const v8h*)(xrow + kk * 32 + 16);   // halves 8..15 : k = 16+hi*8..
        v16h a = __builtin_shufflevector(c0, c1, 0, 1, 2, 3, 4, 5, 6, 7,
                                                 8, 9, 10, 11, 12, 13, 14, 15);
#pragma unroll
        for (int nt = 0; nt < NTN; ++nt) {
            v16h b = *(const v16h*)(wp + (size_t)(kk * NTN + nt) * 512);
            acc[nt] = __builtin_amdgcn_wmma_f32_16x16x32_f16(false, a, false, b,
                                                             (short)0, acc[nt],
                                                             false, false);
        }
    }

#pragma unroll
    for (int nt = 0; nt < NTN; ++nt) {
        int gcol = nt * 16 + lm;
        float bv = bias ? bias[gcol] : 0.f;
#pragma unroll
        for (int r = 0; r < 8; ++r) {
            int grow = mt * 16 + hi * 8 + r;          // D layout: vgpr r
            float v = acc[nt][r] + bv;
            if (Z) v += Z[(size_t)grow * ldz + gcol];
            if (act == 1)      v = v > 0.f ? v : 0.f;
            else if (act == 2) v = v > 0.f ? v : expm1f(v);
            if (Yf32) Yf32[(size_t)grow * ldy32 + gcol] = v;
            if (Yf16) Yf16[(size_t)grow * ldy16 + gcol] = (_Float16)v;
        }
    }
}

// ---------------------------------------------------------------------------
// GATv2 fused edge kernel: one wave per (node, head); lane = feature k (0..31)
// scores via wave32 butterfly reduction; softmax over DEG=16; weighted sum.
// ---------------------------------------------------------------------------
__global__ void gat_edge_kernel(const float* __restrict__ fs,
                                const float* __restrict__ fd,
                                const int* __restrict__ src,
                                const float* __restrict__ attn,
                                const float* __restrict__ bias,
                                float* __restrict__ out, int H)
{
    int lane = threadIdx.x;
    int w = blockIdx.x;
    int n = w / H, hh = w % H;
    int fo = H * 32;
    int col = hh * 32 + lane;

    float ak  = attn[hh * 32 + lane];
    float fdk = fd[(size_t)n * fo + col];

    float fsv[16], sc[16];
#pragma unroll
    for (int d = 0; d < 16; ++d) {
        int row = src[n * 16 + d];
        float v = fs[(size_t)row * fo + col];
        fsv[d] = v;
        float e = v + fdk;
        e = e > 0.f ? e : 0.2f * e;           // leaky_relu(0.2)
        float p = ak * e;
#pragma unroll
        for (int m = 16; m >= 1; m >>= 1) p += __shfl_xor(p, m, 32);
        sc[d] = p;
    }
    float mx = sc[0];
#pragma unroll
    for (int d = 1; d < 16; ++d) mx = fmaxf(mx, sc[d]);
    float s = 0.f;
#pragma unroll
    for (int d = 0; d < 16; ++d) { sc[d] = expf(sc[d] - mx); s += sc[d]; }
    float inv = 1.f / s;
    float o = 0.f;
#pragma unroll
    for (int d = 0; d < 16; ++d) o += sc[d] * fsv[d];
    out[(size_t)n * fo + col] = o * inv + bias[col];
}

// ---------------------------------------------------------------------------
// SAGE LSTM aggregator. One wave handles 16 nodes (independent recurrences).
// pre[N,128] = x@wih.T + b (gate-interleaved, hidden padded to 32).  Per step:
//   - stage 16 gathered rows of pre into LDS via GLOBAL_LOAD_ASYNC_TO_LDS_B128
//     (ASYNCcnt + s_wait_asynccnt) when available
//   - per 16-wide hidden group: 4 WMMAs (i,f,g,o tiles, K=32) with the
//     gathered pre as C operand, immediately folded into the cell update so
//     only 4 accumulators are live (no VGPR spills)
//   - repack h via LDS for the next step's A fragment
// ---------------------------------------------------------------------------
__device__ __forceinline__ float sigmf(float x) { return 1.f / (1.f + expf(-x)); }

__global__ void __launch_bounds__(32, 1)
lstm_kernel(const float* __restrict__ pre,
            const int* __restrict__ src,
            const _Float16* __restrict__ whhP,
            _Float16* __restrict__ hn)
{
    __shared__ __align__(16) float    ldsPre[16 * 128];
    __shared__ __align__(16) _Float16 ldsH[16 * 32];

    int lane = threadIdx.x;
    int hi = lane >> 4, lm = lane & 15;
    int nb = blockIdx.x * 16;

    v16h B[8];
#pragma unroll
    for (int nt = 0; nt < 8; ++nt)
        B[nt] = *(const v16h*)(whhP + ((size_t)nt * 32 + lane) * 16);

    for (int i = lane; i < 512; i += 32) ldsH[i] = (_Float16)0.f;
    v8f c[2] = {{}, {}};
    __syncthreads();

    for (int t = 0; t < 16; ++t) {
        // stage gathered pre rows: 16 rows x 128 f32 (512B/row, 16B per lane)
#if defined(HAVE_ASYNC_LDS)
#pragma unroll
        for (int r = 0; r < 16; ++r) {
            int row = src[(nb + r) * 16 + t];
            const float* gsrc = pre + (size_t)row * 128 + lane * 4;
            float*       ldst = &ldsPre[r * 128 + lane * 4];
            __builtin_amdgcn_global_load_async_to_lds_b128(
                (ASG i32x4*)gsrc, (ASL i32x4*)ldst, 0, 0);
        }
        __builtin_amdgcn_s_wait_asynccnt(0);
#else
#pragma unroll
        for (int r = 0; r < 16; ++r) {
            int row = src[(nb + r) * 16 + t];
            ((float4*)ldsPre)[r * 32 + lane] =
                ((const float4*)(pre + (size_t)row * 128))[lane];
        }
#endif
        __syncthreads();

        // A fragment (16x32 f16) from current hidden state
        v16h a;
#pragma unroll
        for (int h2 = 0; h2 < 16; ++h2) {
            int j = (h2 & 7) + hi * 8 + ((h2 >> 3) << 4);
            a[h2] = ldsH[lm * 32 + j];
        }

        // process hidden groups: tiles [i f g o] per group; 4 live accs only
#pragma unroll
        for (int grp = 0; grp < 2; ++grp) {
            v8f acc[4];
#pragma unroll
            for (int gate = 0; gate < 4; ++gate) {
                int nt = grp * 4 + gate;
                v8f cf;
#pragma unroll
                for (int r = 0; r < 8; ++r)
                    cf[r] = ldsPre[(hi * 8 + r) * 128 + nt * 16 + lm];
                acc[gate] = __builtin_amdgcn_wmma_f32_16x16x32_f16(
                    false, a, false, B[nt], (short)0, cf, false, false);
            }
#pragma unroll
            for (int r = 0; r < 8; ++r) {
                float ig = sigmf(acc[0][r]), fg = sigmf(acc[1][r]);
                float gg = tanhf(acc[2][r]), og = sigmf(acc[3][r]);
                c[grp][r] = fg * c[grp][r] + ig * gg;
                ldsH[(hi * 8 + r) * 32 + grp * 16 + lm] =
                    (_Float16)(og * tanhf(c[grp][r]));
            }
        }
        __syncthreads();
    }

    for (int i = lane; i < 512; i += 32)
        hn[(size_t)nb * 32 + i] = ldsH[i];
}

// ---------------------------------------------------------------------------
// Final MLP: out = relu(z @ f1 + b1) @ f2 + b2     (z: [N,64] f16)
// ---------------------------------------------------------------------------
__global__ void final_mlp_kernel(const _Float16* __restrict__ z,
                                 const float* __restrict__ w1,
                                 const float* __restrict__ b1,
                                 const float* __restrict__ w2,
                                 const float* __restrict__ b2,
                                 float* __restrict__ out, int n)
{
    int tid = blockIdx.x * blockDim.x + threadIdx.x;
    if (tid >= n) return;
    float zz[64];
#pragma unroll
    for (int k = 0; k < 64; ++k) zz[k] = (float)z[(size_t)tid * 64 + k];
    float acc = b2[0];
#pragma unroll 4
    for (int j = 0; j < 16; ++j) {
        float s = b1[j];
        for (int k = 0; k < 64; ++k) s += zz[k] * w1[k * 16 + j];
        acc += fmaxf(s, 0.f) * w2[j];
    }
    out[tid] = acc;
}

// ---------------------------------------------------------------------------
extern "C" void kernel_launch(void* const* d_in, const int* in_sizes, int n_in,
                              void* d_out, int out_size, void* d_ws, size_t ws_size,
                              hipStream_t stream)
{
    (void)in_sizes; (void)n_in; (void)out_size; (void)ws_size;
    const int N = NNODES;

    const float* feat = (const float*)d_in[0];
    const int*   src  = (const int*)d_in[1];
    const float* g_ws[3]   = {(const float*)d_in[2],  (const float*)d_in[10], (const float*)d_in[18]};
    const float* g_bs[3]   = {(const float*)d_in[3],  (const float*)d_in[11], (const float*)d_in[19]};
    const float* g_wd[3]   = {(const float*)d_in[4],  (const float*)d_in[12], (const float*)d_in[20]};
    const float* g_bd[3]   = {(const float*)d_in[5],  (const float*)d_in[13], (const float*)d_in[21]};
    const float* g_attn[3] = {(const float*)d_in[6],  (const float*)d_in[14], (const float*)d_in[22]};
    const float* g_bias[3] = {(const float*)d_in[7],  (const float*)d_in[15], (const float*)d_in[23]};
    const float* l_w[3]    = {(const float*)d_in[8],  (const float*)d_in[16], (const float*)d_in[24]};
    const float* l_b[3]    = {(const float*)d_in[9],  (const float*)d_in[17], (const float*)d_in[25]};
    const float* s_wih[2]    = {(const float*)d_in[26], (const float*)d_in[32]};
    const float* s_whh[2]    = {(const float*)d_in[27], (const float*)d_in[33]};
    const float* s_b[2]      = {(const float*)d_in[28], (const float*)d_in[34]};
    const float* s_wself[2]  = {(const float*)d_in[29], (const float*)d_in[35]};
    const float* s_wneigh[2] = {(const float*)d_in[30], (const float*)d_in[36]};
    const float* s_bias[2]   = {(const float*)d_in[31], (const float*)d_in[37]};
    const float* f1w = (const float*)d_in[38];
    const float* f1b = (const float*)d_in[39];
    const float* f2w = (const float*)d_in[40];
    const float* f2b = (const float*)d_in[41];

    // ---- workspace bump allocator (deterministic layout) ----
    char* ws = (char*)d_ws;
    size_t off = 0;
    auto alloc = [&](size_t bytes) -> char* {
        off = (off + 255) & ~(size_t)255;
        char* p = ws + off;
        off += bytes;
        return p;
    };
    auto packedBytes = [](int Kpad, int Fo) -> size_t {
        return (size_t)(Kpad / 32) * (Fo / 16) * 512 * 2;
    };

    const int gFi[3] = {5, 128, 128}, gFo[3] = {128, 128, 32};
    _Float16 *wsP[3], *wdP[3], *lP[3];
    for (int i = 0; i < 3; ++i) {
        int kp = (gFi[i] + 31) / 32 * 32;
        wsP[i] = (_Float16*)alloc(packedBytes(kp, gFo[i]));
        wdP[i] = (_Float16*)alloc(packedBytes(kp, gFo[i]));
        lP[i]  = (_Float16*)alloc(packedBytes(kp, gFo[i]));
    }
    const int sH[2] = {5, 32};
    _Float16 *ihP[2], *hhP[2], *selfP[2], *neighP[2];
    float* sbP[2];
    for (int j = 0; j < 2; ++j) {
        ihP[j]    = (_Float16*)alloc(packedBytes(32, 128));
        hhP[j]    = (_Float16*)alloc(packedBytes(32, 128));
        selfP[j]  = (_Float16*)alloc(packedBytes(32, 32));
        neighP[j] = (_Float16*)alloc(packedBytes(32, 32));
        sbP[j]    = (float*)alloc(128 * sizeof(float));
    }

    _Float16* featf16 = (_Float16*)alloc((size_t)N * 32 * 2);
    _Float16* hA      = (_Float16*)alloc((size_t)N * 128 * 2);
    _Float16* hB      = (_Float16*)alloc((size_t)N * 128 * 2);
    float*    fsbuf   = (float*)alloc((size_t)N * 128 * 4);
    float*    fdbuf   = (float*)alloc((size_t)N * 128 * 4);
    float*    gatbuf  = (float*)alloc((size_t)N * 128 * 4);
    _Float16* hnb     = (_Float16*)alloc((size_t)N * 32 * 2);
    _Float16* x1      = (_Float16*)alloc((size_t)N * 32 * 2);
    _Float16* zb      = (_Float16*)alloc((size_t)N * 64 * 2);
    float* prebuf = fsbuf;   // GAT path finished before SAGE path uses these
    float* Tbuf   = fdbuf;

    // ---- pack weights ----
    auto packD = [&](const float* W, int K, int Fo, _Float16* Wp) {
        int ntn = Fo / 16, ks = (K + 31) / 32;
        int tot = ks * ntn * 512;
        pack_dense_kernel<<<(tot + 255) / 256, 256, 0, stream>>>(W, K, Fo, Wp, ntn, ks);
    };
    for (int i = 0; i < 3; ++i) {
        packD(g_ws[i], gFi[i], gFo[i], wsP[i]);
        packD(g_wd[i], gFi[i], gFo[i], wdP[i]);
        packD(l_w[i],  gFi[i], gFo[i], lP[i]);
    }
    for (int j = 0; j < 2; ++j) {
        int H = sH[j];
        pack_lstm_kernel<<<16, 256, 0, stream>>>(s_wih[j], H, H, ihP[j]);
        pack_lstm_kernel<<<16, 256, 0, stream>>>(s_whh[j], H, H, hhP[j]);
        pack_lstm_bias_kernel<<<1, 128, 0, stream>>>(s_b[j], H, sbP[j]);
        packD(s_wself[j],  H, 32, selfP[j]);
        packD(s_wneigh[j], H, 32, neighP[j]);
    }

    cvt_pad_kernel<<<(N * 32 + 255) / 256, 256, 0, stream>>>(feat, 5, 32, featf16, N * 32);

    auto gemm = [&](const _Float16* X, int ldx, const _Float16* Wp, const float* bias,
                    const float* Z, int ldz, float* y32, int ly32,
                    _Float16* y16, int ly16, int Fo, int Kpad, int act) {
        int ks = Kpad / 32;
        if (Fo == 128)
            wmma_gemm_strip_kernel<8><<<N / 16, 32, 0, stream>>>(
                X, ldx, Wp, bias, Z, ldz, y32, ly32, y16, ly16, ks, act);
        else
            wmma_gemm_strip_kernel<2><<<N / 16, 32, 0, stream>>>(
                X, ldx, Wp, bias, Z, ldz, y32, ly32, y16, ly16, ks, act);
    };

    // ---- GAT layer 0 (in: feat[ ,5], H=4) ----
    gemm(featf16, 32, wsP[0], g_bs[0], nullptr, 0, fsbuf, 128, nullptr, 0, 128, 32, 0);
    gemm(featf16, 32, wdP[0], g_bd[0], nullptr, 0, fdbuf, 128, nullptr, 0, 128, 32, 0);
    gat_edge_kernel<<<N * 4, 32, 0, stream>>>(fsbuf, fdbuf, src, g_attn[0], g_bias[0], gatbuf, 4);
    gemm(featf16, 32, lP[0], l_b[0], gatbuf, 128, nullptr, 0, hA, 128, 128, 32, 2);

    // ---- GAT layer 1 (in: hA[ ,128], H=4) ----
    gemm(hA, 128, wsP[1], g_bs[1], nullptr, 0, fsbuf, 128, nullptr, 0, 128, 128, 0);
    gemm(hA, 128, wdP[1], g_bd[1], nullptr, 0, fdbuf, 128, nullptr, 0, 128, 128, 0);
    gat_edge_kernel<<<N * 4, 32, 0, stream>>>(fsbuf, fdbuf, src, g_attn[1], g_bias[1], gatbuf, 4);
    gemm(hA, 128, lP[1], l_b[1], gatbuf, 128, nullptr, 0, hB, 128, 128, 128, 2);

    // ---- GAT layer 2 (in: hB[ ,128], H=1; mean over 1 head = identity) ----
    gemm(hB, 128, wsP[2], g_bs[2], nullptr, 0, fsbuf, 32, nullptr, 0, 32, 128, 0);
    gemm(hB, 128, wdP[2], g_bd[2], nullptr, 0, fdbuf, 32, nullptr, 0, 32, 128, 0);
    gat_edge_kernel<<<N * 1, 32, 0, stream>>>(fsbuf, fdbuf, src, g_attn[2], g_bias[2], gatbuf, 1);
    gemm(hB, 128, lP[2], l_b[2], gatbuf, 32, nullptr, 0, zb, 64, 32, 128, 0);   // z[:,0:32]

    // ---- SAGE layer 1 (in: feat) ----
    gemm(featf16, 32, ihP[0], sbP[0], nullptr, 0, prebuf, 128, nullptr, 0, 128, 32, 0);
    lstm_kernel<<<N / 16, 32, 0, stream>>>(prebuf, src, hhP[0], hnb);
    gemm(hnb, 32, neighP[0], nullptr, nullptr, 0, Tbuf, 32, nullptr, 0, 32, 32, 0);
    gemm(featf16, 32, selfP[0], s_bias[0], Tbuf, 32, nullptr, 0, x1, 32, 32, 32, 1);

    // ---- SAGE layer 2 (in: x1) ----
    gemm(x1, 32, ihP[1], sbP[1], nullptr, 0, prebuf, 128, nullptr, 0, 128, 32, 0);
    lstm_kernel<<<N / 16, 32, 0, stream>>>(prebuf, src, hhP[1], hnb);
    gemm(hnb, 32, neighP[1], nullptr, nullptr, 0, Tbuf, 32, nullptr, 0, 32, 32, 0);
    gemm(x1, 32, selfP[1], s_bias[1], Tbuf, 32, nullptr, 0, zb + 32, 64, 32, 32, 1); // z[:,32:64]

    // ---- final MLP ----
    final_mlp_kernel<<<(N + 255) / 256, 256, 0, stream>>>(zb, f1w, f1b, f2w, f2b,
                                                          (float*)d_out, N);
}
